// AxialTemporalDiscriminator_51436528337514
// MI455X (gfx1250) — matrix operand order repaired and checked by reading, not compiled, not run on hardware
//
#include <hip/hip_runtime.h>

// ---------------------------------------------------------------------------
// AxialTemporalDiscriminator for MI455X (gfx1250, wave32, WMMA, async-LDS)
//   B=128, T=20, C=4, HW=4096, D=128, HEADS=8, DH=4, 2 blocks x 2 axes
// ---------------------------------------------------------------------------

typedef __attribute__((ext_vector_type(16))) __bf16 v16bf;
typedef __attribute__((ext_vector_type(8)))  float  v8f;

#define T_LEN 20
#define C_LEN 4
#define D_EMB 128
#define M_ROWS 10240   // 128*20*4
#define EMB_SLAB 8192  // bytes of swizzled W_emb per K-step (8 frags * 1KB)

union Frag16 {            // one WMMA 16-bit operand fragment (per-lane view)
  v16bf v;
  __bf16 e[16];
  uint4 q[2];
};

static __device__ __forceinline__ void st4(float* p, float4 v) {
  p[0] = v.x; p[1] = v.y; p[2] = v.z; p[3] = v.w;
}

// ----- CDNA5 async global->LDS copy (guarded; falls back to sync staging) --
#if defined(__AMDGCN__) && __has_builtin(__builtin_amdgcn_global_load_async_to_lds_b128)
#define ASYNC_LDS 1
#else
#define ASYNC_LDS 0
#endif

typedef int vsi4 __attribute__((vector_size(16)));             // 4 x i32
typedef __attribute__((address_space(1))) vsi4* gas1_t;        // global ptr
typedef __attribute__((address_space(3))) vsi4* las3_t;        // LDS ptr

static __device__ __forceinline__ void copy16_g2l(void* lds, const void* g) {
#if ASYNC_LDS
  __builtin_amdgcn_global_load_async_to_lds_b128(
      (gas1_t)(unsigned long long)g,
      (las3_t)(unsigned int)(unsigned long long)lds, 0, 0);
#else
  *(uint4*)lds = *(const uint4*)g;
#endif
}

static __device__ __forceinline__ void copy_fence() {
#if ASYNC_LDS
#if __has_builtin(__builtin_amdgcn_s_wait_asynccnt)
  __builtin_amdgcn_s_wait_asynccnt(0);
#else
  asm volatile("s_wait_asynccnt 0x0" ::: "memory");
#endif
#endif
}

// ---------------------------------------------------------------------------
// Prep kernels: pre-swizzle B operands into per-lane fragment order.
// B (32x16 bf16 tile, wave32): lane 0..15 -> col n=lane, K=kblk*32+0..15;
// lane 16..31 -> col n=lane-16, K=kblk*32+16..31. Flat [kblk][nblk][lane][16].
// ---------------------------------------------------------------------------

__global__ void prep_swizzle_emb(const float* __restrict__ W,      // [4096,128]
                                 __bf16* __restrict__ dst) {
  int idx = blockIdx.x * 256 + threadIdx.x;   // < 4096*128
  int frag = idx >> 9, within = idx & 511;
  int lane = within >> 4, j = within & 15;
  int kblk = frag >> 3, nblk = frag & 7;      // Nf = 128/16 = 8
  int k = kblk * 32 + ((lane >> 4) << 4) + j;
  int n = nblk * 16 + (lane & 15);
  dst[idx] = (__bf16)W[(size_t)k * 128 + n];
}

// Fused [Wq_T | Wkv_T | Wq_C | Wkv_C] -> [128,192] per block, swizzled.
__global__ void prep_swizzle_qkv(const float* __restrict__ Wq,   // [2,2,128,32]
                                 const float* __restrict__ Wkv,  // [2,2,128,64]
                                 __bf16* __restrict__ dst) {
  int i = blockIdx.y;
  int idx = blockIdx.x * 256 + threadIdx.x;    // < 128*192
  int frag = idx >> 9, within = idx & 511;
  int lane = within >> 4, j = within & 15;
  int kblk = frag / 12, nblk = frag % 12;      // Nf = 192/16 = 12
  int k = kblk * 32 + ((lane >> 4) << 4) + j;
  int n = nblk * 16 + (lane & 15);
  float v;
  if (n < 32)       v = Wq [((size_t)(i*2+0)*128 + k)*32 + n];
  else if (n < 96)  v = Wkv[((size_t)(i*2+0)*128 + k)*64 + (n-32)];
  else if (n < 128) v = Wq [((size_t)(i*2+1)*128 + k)*32 + (n-96)];
  else              v = Wkv[((size_t)(i*2+1)*128 + k)*64 + (n-128)];
  dst[(size_t)i * 128 * 192 + idx] = (__bf16)v;
}

// Fused [Wout_T ; Wout_C] -> [64,128] per block, swizzled. Also bias sums.
__global__ void prep_swizzle_out(const float* __restrict__ Wout,  // [2,2,32,128]
                                 const float* __restrict__ bout,  // [2,2,128]
                                 __bf16* __restrict__ dst,
                                 float* __restrict__ biassum) {
  int i = blockIdx.y;
  int idx = blockIdx.x * 256 + threadIdx.x;    // < 64*128
  int frag = idx >> 9, within = idx & 511;
  int lane = within >> 4, j = within & 15;
  int kblk = frag >> 3, nblk = frag & 7;       // Nf = 128/16 = 8
  int k = kblk * 32 + ((lane >> 4) << 4) + j;
  int n = nblk * 16 + (lane & 15);
  float v = (k < 32) ? Wout[((size_t)(i*2+0)*32 + k     )*128 + n]
                     : Wout[((size_t)(i*2+1)*32 + (k-32))*128 + n];
  dst[(size_t)i * 64 * 128 + idx] = (__bf16)v;
  if (blockIdx.x == 0 && threadIdx.x < 128)
    biassum[i*128 + threadIdx.x] =
        bout[(i*2+0)*128 + threadIdx.x] + bout[(i*2+1)*128 + threadIdx.x];
}

// ---------------------------------------------------------------------------
// Embed GEMM: h = leaky((x + 1e-3 n1 + 1e-4 n2) @ W_emb + b), bf16 WMMA.
// 8 waves/block, each owns 16 rows x full 128 cols (8 accumulators), so the
// 503 MB input stream is read exactly once (HBM roofline floor ~21.6 us).
// The 8 KB/K-step B slab is shared by the block: async-copied global->LDS,
// double-buffered, cutting B L2 traffic 8x.
// ---------------------------------------------------------------------------
__global__ __launch_bounds__(256)
void embed_gemm(const float* __restrict__ x, const float* __restrict__ n1,
                const float* __restrict__ n2,
                const __bf16* __restrict__ Bf,
                const float* __restrict__ bias,
                float* __restrict__ C, __bf16* __restrict__ Cbf) {
  __shared__ __align__(16) unsigned char bsm[2 * EMB_SLAB];
  const int tid  = threadIdx.x;
  const int wave = tid >> 5;
  const int lane = tid & 31;
  const int half = lane >> 4;
  const int mtile = blockIdx.x * 8 + wave;          // 640 tiles
  const int mrow = mtile * 16 + (lane & 15);
  const size_t rowoff = (size_t)mrow * 4096;

  const v8f vz = {0.f,0.f,0.f,0.f,0.f,0.f,0.f,0.f};
  v8f acc[8];
#pragma unroll
  for (int i = 0; i < 8; ++i) acc[i] = vz;

  // kick off slab 0
  {
    const unsigned char* src = (const unsigned char*)Bf;
    copy16_g2l(bsm + tid * 16, src + tid * 16);
    copy16_g2l(bsm + 4096 + tid * 16, src + 4096 + tid * 16);
  }

  for (int ks = 0; ks < 128; ++ks) {                // K = 4096 in steps of 32
    const int buf = ks & 1;
    copy_fence();            // own async portion of slab `ks` landed in LDS
    __syncthreads();         // whole slab visible; prev buffer fully consumed
    if (ks + 1 < 128) {
      const unsigned char* src =
          (const unsigned char*)Bf + (size_t)(ks + 1) * EMB_SLAB;
      unsigned char* dst = bsm + (buf ^ 1) * EMB_SLAB;
      copy16_g2l(dst + tid * 16, src + tid * 16);
      copy16_g2l(dst + 4096 + tid * 16, src + 4096 + tid * 16);
    }

    const int kh = ks * 32 + half * 16;
    // prefetch the streams ~64 K-steps ahead (speculative, fault-free)
    __builtin_prefetch((const void*)(x  + rowoff + kh + 2048), 0, 0);
    __builtin_prefetch((const void*)(n1 + rowoff + kh + 2048), 0, 0);
    __builtin_prefetch((const void*)(n2 + rowoff + kh + 2048), 0, 0);

    Frag16 a;
#pragma unroll
    for (int ch = 0; ch < 2; ++ch) {
      const size_t base = rowoff + (size_t)(kh + 8 * ch);
      float fx[8], fa[8], fb[8];
      st4(fx,     *(const float4*)(x  + base));
      st4(fx + 4, *(const float4*)(x  + base + 4));
      st4(fa,     *(const float4*)(n1 + base));
      st4(fa + 4, *(const float4*)(n1 + base + 4));
      st4(fb,     *(const float4*)(n2 + base));
      st4(fb + 4, *(const float4*)(n2 + base + 4));
#pragma unroll
      for (int j = 0; j < 8; ++j)
        a.e[ch * 8 + j] = (__bf16)(fx[j] + 1e-3f * fa[j] + 1e-4f * fb[j]);
    }

    const uint4* bb = (const uint4*)(bsm + buf * EMB_SLAB) + lane * 2;
#pragma unroll
    for (int nb = 0; nb < 8; ++nb) {
      Frag16 b;
      b.q[0] = bb[nb * 64 + 0];
      b.q[1] = bb[nb * 64 + 1];
      acc[nb] = __builtin_amdgcn_wmma_f32_16x16x32_bf16(
          false, a.v, false, b.v, (short)0, acc[nb], false, false);
    }
  }

#pragma unroll
  for (int nb = 0; nb < 8; ++nb) {
#pragma unroll
    for (int r = 0; r < 8; ++r) {
      int m = mtile * 16 + r + 8 * half;
      int n = nb * 16 + (lane & 15);
      float v = acc[nb][r] + bias[n];
      v = v > 0.f ? v : 0.2f * v;                   // leaky 0.2
      C  [(size_t)m * D_EMB + n] = v;
      Cbf[(size_t)m * D_EMB + n] = (__bf16)v;
    }
  }
}

// ---------------------------------------------------------------------------
// Generic bf16 WMMA GEMM: C[M x N] = A[M x K] @ Bfrag + (bias), opt. leaky,
// optional bf16 mirror. One wave computes 16 x 64 (grid.y tiles N).
// ---------------------------------------------------------------------------
__global__ __launch_bounds__(256)
void gemm_bf16(const __bf16* __restrict__ A, int lda,
               const __bf16* __restrict__ Bf, int Nf, int nK,
               float* __restrict__ C, int ldc,
               const float* __restrict__ bias, int leaky,
               __bf16* __restrict__ Cbf, int M) {
  const int wave = threadIdx.x >> 5;
  const int lane = threadIdx.x & 31;
  const int half = lane >> 4;
  const int mtile = blockIdx.x * 8 + wave;
  if (mtile * 16 >= M) return;
  const int ntile = blockIdx.y * 4;                 // 4 col-tiles = 64 cols
  const __bf16* Arow = A + (size_t)(mtile * 16 + (lane & 15)) * lda;

  const v8f vz = {0.f,0.f,0.f,0.f,0.f,0.f,0.f,0.f};
  v8f acc[4];
#pragma unroll
  for (int i = 0; i < 4; ++i) acc[i] = vz;

  for (int ks = 0; ks < nK; ++ks) {
    const int kh = ks * 32 + half * 16;
    Frag16 a;
    a.q[0] = *(const uint4*)(Arow + kh);
    a.q[1] = *(const uint4*)(Arow + kh + 8);
#pragma unroll
    for (int nb = 0; nb < 4; ++nb) {
      const int frag = ks * Nf + ntile + nb;
      const uint4* bp = (const uint4*)Bf + (size_t)frag * 64 + lane * 2;
      Frag16 b;
      b.q[0] = bp[0];
      b.q[1] = bp[1];
      acc[nb] = __builtin_amdgcn_wmma_f32_16x16x32_bf16(
          false, a.v, false, b.v, (short)0, acc[nb], false, false);
    }
  }

#pragma unroll
  for (int nb = 0; nb < 4; ++nb) {
#pragma unroll
    for (int r = 0; r < 8; ++r) {
      int m = mtile * 16 + r + 8 * half;
      int n = (ntile + nb) * 16 + (lane & 15);
      float v = acc[nb][r];
      if (bias) v += bias[n];
      if (leaky) v = v > 0.f ? v : 0.2f * v;
      C[(size_t)m * ldc + n] = v;
      if (Cbf) Cbf[(size_t)m * ldc + n] = (__bf16)v;
    }
  }
}

// ---------------------------------------------------------------------------
// Softmax cores. qkv layout per row (b,t,c), 192 f32 cols:
//   [0:32) qT | [32:64) kT | [64:96) vT | [96:128) qC | [128:160) kC | [160:192) vC
// o_all per row: 64 bf16 cols: [0:32) o_T | [32:64) o_C  (head-major)
// ---------------------------------------------------------------------------
__global__ __launch_bounds__(256)
void attn_axis_t(const float* __restrict__ qkv, __bf16* __restrict__ o_all) {
  __shared__ float kv[T_LEN * 64];                  // k(32) | v(32) per t
  const int bc = blockIdx.x;                        // 512 = B*C
  const int b = bc >> 2, c = bc & 3;
  for (int idx = threadIdx.x; idx < T_LEN * 64; idx += 256) {
    int t = idx >> 6, j = idx & 63;
    int grow = (b * T_LEN + t) * C_LEN + c;
    kv[idx] = qkv[(size_t)grow * 192 + 32 + j];
  }
  __syncthreads();
  if (threadIdx.x < 8 * T_LEN) {
    const int hh = threadIdx.x & 7, t = threadIdx.x >> 3;
    const int grow = (b * T_LEN + t) * C_LEN + c;
    const float* qp = qkv + (size_t)grow * 192 + hh * 4;
    float q0 = qp[0], q1 = qp[1], q2 = qp[2], q3 = qp[3];
    float s[T_LEN], mx = -3.4e38f;
#pragma unroll
    for (int k2 = 0; k2 < T_LEN; ++k2) {
      const float* kk = kv + k2 * 64 + hh * 4;
      float d = q0*kk[0] + q1*kk[1] + q2*kk[2] + q3*kk[3];
      s[k2] = d * 0.5f;                             // DH^-0.5
      mx = fmaxf(mx, s[k2]);
    }
    float den = 0.f;
#pragma unroll
    for (int k2 = 0; k2 < T_LEN; ++k2) { s[k2] = __expf(s[k2] - mx); den += s[k2]; }
    float o0 = 0, o1 = 0, o2 = 0, o3 = 0;
#pragma unroll
    for (int k2 = 0; k2 < T_LEN; ++k2) {
      const float* vv = kv + k2 * 64 + 32 + hh * 4;
      o0 += s[k2]*vv[0]; o1 += s[k2]*vv[1]; o2 += s[k2]*vv[2]; o3 += s[k2]*vv[3];
    }
    float inv = 1.f / den;
    __bf16* op = o_all + (size_t)grow * 64 + hh * 4;
    op[0] = (__bf16)(o0*inv); op[1] = (__bf16)(o1*inv);
    op[2] = (__bf16)(o2*inv); op[3] = (__bf16)(o3*inv);
  }
}

__global__ __launch_bounds__(256)
void attn_axis_c(const float* __restrict__ qkv, __bf16* __restrict__ o_all) {
  __shared__ float kv[8][C_LEN * 64];
  const int p = threadIdx.x >> 5, w = threadIdx.x & 31;
  const int bt = blockIdx.x * 8 + p;                // 2560 = B*T
  const int b = bt / T_LEN, t = bt % T_LEN;
  for (int idx = w; idx < C_LEN * 64; idx += 32) {
    int c = idx >> 6, j = idx & 63;
    int grow = (b * T_LEN + t) * C_LEN + c;
    kv[p][idx] = qkv[(size_t)grow * 192 + 128 + j];
  }
  __syncthreads();
  const int qi = w >> 3, hh = w & 7;                // 4 queries x 8 heads
  const int grow = (b * T_LEN + t) * C_LEN + qi;
  const float* qp = qkv + (size_t)grow * 192 + 96 + hh * 4;
  float q0 = qp[0], q1 = qp[1], q2 = qp[2], q3 = qp[3];
  float s[C_LEN], mx = -3.4e38f;
#pragma unroll
  for (int k2 = 0; k2 < C_LEN; ++k2) {
    const float* kk = kv[p] + k2 * 64 + hh * 4;
    float d = q0*kk[0] + q1*kk[1] + q2*kk[2] + q3*kk[3];
    s[k2] = d * 0.5f;
    mx = fmaxf(mx, s[k2]);
  }
  float den = 0.f;
#pragma unroll
  for (int k2 = 0; k2 < C_LEN; ++k2) { s[k2] = __expf(s[k2] - mx); den += s[k2]; }
  float o0 = 0, o1 = 0, o2 = 0, o3 = 0;
#pragma unroll
  for (int k2 = 0; k2 < C_LEN; ++k2) {
    const float* vv = kv[p] + k2 * 64 + 32 + hh * 4;
    o0 += s[k2]*vv[0]; o1 += s[k2]*vv[1]; o2 += s[k2]*vv[2]; o3 += s[k2]*vv[3];
  }
  float inv = 1.f / den;
  __bf16* op = o_all + (size_t)grow * 64 + 32 + hh * 4;
  op[0] = (__bf16)(o0*inv); op[1] = (__bf16)(o1*inv);
  op[2] = (__bf16)(o2*inv); op[3] = (__bf16)(o3*inv);
}

// ---------------------------------------------------------------------------
// Decoder: out[b] = sigmoid(h[b,:].dot(W_dec) + b_dec)
// ---------------------------------------------------------------------------
__global__ __launch_bounds__(256)
void decoder(const float* __restrict__ h, const float* __restrict__ Wd,
             const float* __restrict__ bd, float* __restrict__ out) {
  __shared__ float red[256];
  const int b = blockIdx.x;
  float p = 0.f;
  for (int idx = threadIdx.x; idx < 10240; idx += 256)
    p += h[(size_t)b * 10240 + idx] * Wd[idx];
  red[threadIdx.x] = p;
  __syncthreads();
  for (int s = 128; s > 0; s >>= 1) {
    if (threadIdx.x < s) red[threadIdx.x] += red[threadIdx.x + s];
    __syncthreads();
  }
  if (threadIdx.x == 0)
    out[b] = 1.f / (1.f + __expf(-(red[0] + bd[0])));
}

// ---------------------------------------------------------------------------
extern "C" void kernel_launch(void* const* d_in, const int* in_sizes, int n_in,
                              void* d_out, int out_size, void* d_ws, size_t ws_size,
                              hipStream_t stream) {
  (void)in_sizes; (void)n_in; (void)out_size; (void)ws_size;
  const float* x     = (const float*)d_in[0];
  const float* n1    = (const float*)d_in[1];
  const float* n2    = (const float*)d_in[2];
  const float* W_emb = (const float*)d_in[3];
  const float* b_emb = (const float*)d_in[4];
  const float* Wq    = (const float*)d_in[5];
  const float* Wkv   = (const float*)d_in[6];
  const float* Wout  = (const float*)d_in[7];
  const float* bout  = (const float*)d_in[8];
  const float* W_dec = (const float*)d_in[9];
  const float* b_dec = (const float*)d_in[10];

  char* W = (char*)d_ws;                       // ~18.3 MB total, 256B-aligned
  float*   h_f    = (float*) (W + 0);          // 10240*128*4
  __bf16*  h_bf   = (__bf16*)(W + 5242880);    // 10240*128*2
  __bf16*  wemb_f = (__bf16*)(W + 7864320);    // 4096*128*2
  __bf16*  bqkv_f = (__bf16*)(W + 8912896);    // 2*128*192*2
  __bf16*  bout_f = (__bf16*)(W + 9011200);    // 2*64*128*2
  float*   bsum   = (float*) (W + 9043968);    // 2*128*4
  float*   qkvbuf = (float*) (W + 9044992);    // 10240*192*4
  __bf16*  o_all  = (__bf16*)(W + 16909312);   // 10240*64*2

  prep_swizzle_emb<<<2048, 256, 0, stream>>>(W_emb, wemb_f);
  prep_swizzle_qkv<<<dim3(96, 2), 256, 0, stream>>>(Wq, Wkv, bqkv_f);
  prep_swizzle_out<<<dim3(32, 2), 256, 0, stream>>>(Wout, bout, bout_f, bsum);

  embed_gemm<<<80, 256, 0, stream>>>(x, n1, n2, wemb_f, b_emb, h_f, h_bf);

  for (int i = 0; i < 2; ++i) {
    gemm_bf16<<<dim3(80, 3), 256, 0, stream>>>(
        h_bf, 128, bqkv_f + (size_t)i * 128 * 192, 12, 4,
        qkvbuf, 192, nullptr, 0, nullptr, M_ROWS);
    attn_axis_t<<<512, 256, 0, stream>>>(qkvbuf, o_all);
    attn_axis_c<<<320, 256, 0, stream>>>(qkvbuf, o_all);
    gemm_bf16<<<dim3(80, 2), 256, 0, stream>>>(
        o_all, 64, bout_f + (size_t)i * 64 * 128, 8, 2,
        h_f, 128, bsum + i * 128, 1, h_bf, M_ROWS);
  }

  decoder<<<128, 256, 0, stream>>>(h_f, W_dec, b_dec, (float*)d_out);
}
